// LSTM_head_14216341750112
// MI455X (gfx1250) — matrix operand order repaired and checked by reading, not compiled, not run on hardware
//
#include <hip/hip_runtime.h>
#include <hip/hip_bf16.h>
#include <stdint.h>

// Problem constants (match reference)
#define BQ 256
#define TT 512
#define EE 512
#define HH 512
#define G4 2048   // 4*H
#define KQ 1024   // E + H

typedef _Float16 v16h __attribute__((ext_vector_type(16)));
typedef _Float16 v8h  __attribute__((ext_vector_type(8)));
typedef float    v8f  __attribute__((ext_vector_type(8)));
typedef unsigned int u32x4 __attribute__((ext_vector_type(4)));
typedef int  i32x4 __attribute__((ext_vector_type(4)));
typedef int  i32x8 __attribute__((ext_vector_type(8)));

// Toolchain detection: therock-10.0 headers ship the TDM descriptor header and
// use the 6-arg builtin; ROCm 7.2 uses the 5-arg form.
#if __has_include(<hip/amd_detail/amd_gfx1250_TDM.h>)
#define TDM_6ARG 1
#else
#define TDM_6ARG 0
#endif

// ---------------------------------------------------------------------------
// Tensor Data Mover: 2D tile load (global -> LDS), D# packed per ISA 08 §8.
// group0: [1:0]=count=1, [63:32]=lds_addr, [120:64]=global_addr, [127:126]=type(2)
// group1: [17:16]=data_size, [79:48]=tensor_dim0, [111:80]=tensor_dim1,
//         [127:112]=tile_dim0, [143:128]=tile_dim1, [207:160]=tensor_dim0_stride
// ---------------------------------------------------------------------------
__device__ __forceinline__ void tdm_load_2d(unsigned lds_byte_addr,
                                            unsigned long long gaddr,
                                            unsigned dsz_code,
                                            unsigned tile_d0, unsigned tile_d1,
                                            unsigned tensor_d0, unsigned tensor_d1,
                                            unsigned long long stride0_elems)
{
  u32x4 g0;
  g0.x = 1u;                                                  // count=1, user desc
  g0.y = lds_byte_addr;
  g0.z = (unsigned)(gaddr & 0xFFFFFFFFull);
  g0.w = (unsigned)((gaddr >> 32) & 0x01FFFFFFull) | (2u << 30); // type = 2 (image)
  i32x8 g1;
  g1[0] = (int)(dsz_code << 16);                              // wg_mask=0, data_size
  g1[1] = (int)((tensor_d0 & 0xFFFFu) << 16);
  g1[2] = (int)(((tensor_d0 >> 16) & 0xFFFFu) | ((tensor_d1 & 0xFFFFu) << 16));
  g1[3] = (int)(((tensor_d1 >> 16) & 0xFFFFu) | ((tile_d0 & 0xFFFFu) << 16));
  g1[4] = (int)(tile_d1 & 0xFFFFu);                           // tile_dim2 = 0
  g1[5] = (int)(stride0_elems & 0xFFFFFFFFull);
  g1[6] = (int)((stride0_elems >> 32) & 0xFFFFull);           // dim1_stride unused (2D)
  g1[7] = 0;
  i32x4 z4 = (i32x4)0;
#if TDM_6ARG
  i32x8 z8 = (i32x8)0;
  __builtin_amdgcn_tensor_load_to_lds(g0, g1, z4, z4, z8, 0);
#else
  __builtin_amdgcn_tensor_load_to_lds(g0, g1, z4, z4, 0);
#endif
}

__device__ __forceinline__ float fast_sigmoid(float x) {
  return 1.0f / (1.0f + __expf(-x));
}
__device__ __forceinline__ float fast_tanh(float x) {
  x = fminf(fmaxf(x, -15.0f), 15.0f);
  float e = __expf(2.0f * x);
  return (e - 1.0f) / (e + 1.0f);
}

// Relaxed spin until *p >= want, then one acquire fence (avoids a WGP-cache
// invalidate per poll iteration).
__device__ __forceinline__ void spin_ge(const int* p, int n, int want) {
  #pragma unroll 1
  for (int j = 0; j < n; ++j) {
    while (__hip_atomic_load(p + j, __ATOMIC_RELAXED, __HIP_MEMORY_SCOPE_AGENT) < want)
      __builtin_amdgcn_s_sleep(1);
  }
  __builtin_amdgcn_fence(__ATOMIC_ACQUIRE, "agent");
}

// LDS layout (bytes), total = 320 KB exactly
#define LDS_W_OFF   0u        // 128 x 1024 f16  = 262144
#define LDS_AX_OFF  262144u   // 16 x 512  f16   = 16384   (x part of A)
#define LDS_AH_OFF  278528u   // 16 x 512  f16   = 16384   (h part of A)
#define LDS_XS_OFF  294912u   // 16 x 512  f32   = 32768   (x staging; aliases gate exch)
#define LDS_BYTES   327680u

// ---------------------------------------------------------------------------
// Prep: Wcat[g][0:512]=W_ih[g], Wcat[g][512:1024]=W_hh[g] (f16); bias=b_ih+b_hh;
// zero h double-buffer; init flags. Re-run each launch => replay-deterministic.
// ---------------------------------------------------------------------------
__global__ void prep_kernel(const float* __restrict__ W_ih, const float* __restrict__ W_hh,
                            const float* __restrict__ b_ih, const float* __restrict__ b_hh,
                            _Float16* __restrict__ Wcat, float* __restrict__ bias,
                            _Float16* __restrict__ hbuf, int* __restrict__ done,
                            int* __restrict__ rdone)
{
  const int bid = blockIdx.x, tid = threadIdx.x;
  if (bid < 2048) {
    for (int e = tid; e < 512; e += 256) {
      Wcat[bid * 1024 + e]       = (_Float16)W_ih[bid * 512 + e];
      Wcat[bid * 1024 + 512 + e] = (_Float16)W_hh[bid * 512 + e];
    }
  } else if (bid == 2048) {
    for (int i = tid; i < 2048; i += 256) bias[i] = b_ih[i] + b_hh[i];
  } else if (bid < 2048 + 1 + 128) {
    const int base = (bid - 2049) * 2048;   // 128 blocks * 2048 = 2*256*512 f16
    for (int j = tid; j < 2048; j += 256) hbuf[base + j] = (_Float16)0.0f;
  } else {
    if (tid < 256) { done[tid] = 0; rdone[tid] = -1; }
  }
}

// ---------------------------------------------------------------------------
// Persistent fused LSTM. Grid = 256 blocks = 16 batch-groups x 16 col-groups.
// Block (bg,cg): owns batch rows bg*16..+16, h columns cg*32..+32.
// Wave w (of 4) computes gate w (i/f/g/o order) as two 16x16 WMMA N-tiles,
// K = 1024 = [x_t | h_t].
// ---------------------------------------------------------------------------
__global__ void __launch_bounds__(128)
lstm_kernel(const float* __restrict__ embeds,
            const _Float16* __restrict__ Wcat,
            const float* __restrict__ bias,
            _Float16* __restrict__ hbuf,   // two phases of [256][512] f16
            int* __restrict__ done, int* __restrict__ rdone)
{
  extern __shared__ __align__(16) unsigned char smem[];
  _Float16* Wl   = (_Float16*)(smem + LDS_W_OFF);   // [128][1024]
  _Float16* Ax   = (_Float16*)(smem + LDS_AX_OFF);  // [16][512]
  _Float16* Ah   = (_Float16*)(smem + LDS_AH_OFF);  // [16][512]
  float*    Xs   = (float*)(smem + LDS_XS_OFF);     // [16][512] f32 staging
  float*    Exch = Xs;                              // gate exchange aliases staging

  const int tid  = threadIdx.x;
  const int lane = tid & 31;
  const int wave = tid >> 5;        // 0..3 == gate index (i,f,g,o)
  const int nlo  = lane & 15;
  const int half = lane >> 4;
  const int bg   = blockIdx.x >> 4;
  const int cg   = blockIdx.x & 15;
  const int fbase = bg * 16;        // first batch row owned
  const unsigned HB = (unsigned)BQ * (unsigned)HH;  // elems per h phase

  // --- one-time TDM load of this block's 128x1024 f16 weight slice into LDS ---
  // 4 gate groups of 32 contiguous rows, inter-group stride 512 rows.
  if (tid < 32) {
    unsigned long long wg =
        (unsigned long long)(uintptr_t)(Wcat + (unsigned long long)cg * 32ull * 1024ull);
    tdm_load_2d((unsigned)(uintptr_t)Wl, wg, /*f16*/1u,
                /*tile*/32768u, 4u, /*tensor*/32768u, 4u, /*stride*/524288ull);
    __builtin_amdgcn_s_wait_tensorcnt(0);
  }
  __syncthreads();

  // Per-lane bias for the two owned N tiles (bias is per-column, same for all M)
  const float bias0 = bias[wave * 512 + cg * 32 + 0 * 16 + nlo];
  const float bias1 = bias[wave * 512 + cg * 32 + 1 * 16 + nlo];

  float creg[8];                      // c state, lane-local (waves 0,1 use it)
  #pragma unroll
  for (int v = 0; v < 8; ++v) creg[v] = 0.0f;

  for (int t = 0; t < TT; ++t) {
    // (1) x_t has no cross-block dependency: start its DMA first so it overlaps
    //     the spin-wait on peer h_t flags.
    if (tid < 32) {
      unsigned long long xg = (unsigned long long)(uintptr_t)
          (embeds + ((unsigned long long)fbase * TT + (unsigned long long)t) * EE);
      tdm_load_2d((unsigned)(uintptr_t)Xs, xg, /*f32*/2u, 512u, 16u, 512u, 16u,
                  (unsigned long long)(TT * EE));
    }

    // (2) wait until every peer col-group published h_t, then DMA the h tile
    if (tid == 0) spin_ge(done + bg * 16, 16, t);
    if (tid < 32) {
      unsigned long long hg = (unsigned long long)(uintptr_t)
          (hbuf + (unsigned long long)(t & 1) * HB + (unsigned long long)fbase * HH);
      tdm_load_2d((unsigned)(uintptr_t)Ah, hg, /*f16*/1u, 512u, 16u, 512u, 16u, 512ull);
      // TENSORcnt completes in issue order: <=1 outstanding means x is done.
      __builtin_amdgcn_s_wait_tensorcnt(1);
    }
    __syncthreads();

    // (3) convert x staging f32 -> f16 A_x plane (h tile still in flight)
    #pragma unroll 4
    for (int idx = tid; idx < 16 * 512; idx += 128)
      Ax[idx] = (_Float16)Xs[idx];
    if (tid < 32) __builtin_amdgcn_s_wait_tensorcnt(0);   // h tile landed
    if (tid == 0)   // publish "read of phase t&1 done" (anti-WAR handshake)
      __hip_atomic_store(rdone + bg * 16 + cg, t, __ATOMIC_RELEASE,
                         __HIP_MEMORY_SCOPE_AGENT);
    __syncthreads();

    // (4) GEMM: gates[16 x 32] for gate `wave`, K loop over 32 windows of 32
    v8f acc0, acc1;
    #pragma unroll
    for (int v = 0; v < 8; ++v) { acc0[v] = bias0; acc1[v] = bias1; }

    const int m   = nlo;                      // A row handled by this lane
    const int wr0 = wave * 32 + 0 * 16 + nlo; // B (weight) rows for N tiles
    const int wr1 = wave * 32 + 1 * 16 + nlo;

    #pragma unroll 4
    for (int k = 0; k < 32; ++k) {
      const _Float16* Ap = (k < 16) ? Ax : Ah;
      const int kk = (k & 15) * 32;
      // A fragment, 16x32 f16: lanes 0-15 hold K 0-15, lanes 16-31 hold K 16-31
      v16h a = *(const v16h*)(Ap + m * 512 + kk + half * 16);
      // B fragment, 32x16 f16: lanes 0-15 hold K 0-15, lanes 16-31 hold K 16-31
      v16h b0 = *(const v16h*)(Wl + wr0 * 1024 + k * 32 + half * 16);
      v16h b1 = *(const v16h*)(Wl + wr1 * 1024 + k * 32 + half * 16);
      acc0 = __builtin_amdgcn_wmma_f32_16x16x32_f16(false, a, false, b0,
                                                    (short)0, acc0, false, false);
      acc1 = __builtin_amdgcn_wmma_f32_16x16x32_f16(false, a, false, b1,
                                                    (short)0, acc1, false, false);
    }

    // (5) publish gate tiles to LDS exchange (C layout: M = v+8*half, N = nlo)
    #pragma unroll
    for (int v = 0; v < 8; ++v) {
      const int mm = v + 8 * half;
      Exch[(wave * 2 + 0) * 256 + mm * 16 + nlo] = acc0[v];
      Exch[(wave * 2 + 1) * 256 + mm * 16 + nlo] = acc1[v];
    }
    __syncthreads();

    // Anti-WAR: h_{t+1} overwrites the phase peers read at step t-1
    if (tid == 0) spin_ge(rdone + bg * 16, 16, t - 1);
    __syncthreads();

    // (6) lane-local LSTM cell update + h_{t+1} store (waves 0,1 each own 16 cols)
    if (wave < 2) {
      _Float16* hout = hbuf + (unsigned long long)((t + 1) & 1) * HB;
      #pragma unroll
      for (int v = 0; v < 8; ++v) {
        const int mm = v + 8 * half;
        const int idx = mm * 16 + nlo;
        const float gi = Exch[(0 * 2 + wave) * 256 + idx];
        const float gf = Exch[(1 * 2 + wave) * 256 + idx];
        const float gg = Exch[(2 * 2 + wave) * 256 + idx];
        const float go = Exch[(3 * 2 + wave) * 256 + idx];
        const float c  = fast_sigmoid(gf) * creg[v] + fast_sigmoid(gi) * fast_tanh(gg);
        creg[v] = c;
        const float h = fast_sigmoid(go) * fast_tanh(c);
        hout[(unsigned long long)(fbase + mm) * HH + cg * 32 + wave * 16 + nlo] =
            (_Float16)h;
      }
      __threadfence();   // release h stores to device scope before flagging
    }
    __syncthreads();
    if (tid == 0)
      __hip_atomic_store(done + bg * 16 + cg, t + 1, __ATOMIC_RELEASE,
                         __HIP_MEMORY_SCOPE_AGENT);
  }
}

// ---------------------------------------------------------------------------
// Head: out = relu(h @ fc1^T + b1) @ fc2^T + b2   (tiny: 256x3)
// ---------------------------------------------------------------------------
__global__ void head_kernel(const _Float16* __restrict__ hfin,
                            const float* __restrict__ fc1_w, const float* __restrict__ fc1_b,
                            const float* __restrict__ fc2_w, const float* __restrict__ fc2_b,
                            float* __restrict__ out)
{
  __shared__ float hrow[512];
  __shared__ float x1[512];
  const int b = blockIdx.x, tid = threadIdx.x;
  hrow[tid]       = (float)hfin[b * 512 + tid];
  hrow[tid + 256] = (float)hfin[b * 512 + tid + 256];
  __syncthreads();
  for (int oc = tid; oc < 512; oc += 256) {
    float acc = fc1_b[oc];
    const float* w = fc1_w + oc * 512;
    #pragma unroll 8
    for (int k = 0; k < 512; ++k) acc += w[k] * hrow[k];
    x1[oc] = fmaxf(acc, 0.0f);
  }
  __syncthreads();
  if (tid < 3) {
    float acc = fc2_b[tid];
    const float* w = fc2_w + tid * 512;
    for (int k = 0; k < 512; ++k) acc += w[k] * x1[k];
    out[b * 3 + tid] = acc;
  }
}

// ---------------------------------------------------------------------------
extern "C" void kernel_launch(void* const* d_in, const int* in_sizes, int n_in,
                              void* d_out, int out_size, void* d_ws, size_t ws_size,
                              hipStream_t stream)
{
  const float* embeds = (const float*)d_in[0];
  const float* W_ih   = (const float*)d_in[1];
  const float* W_hh   = (const float*)d_in[2];
  const float* b_ih   = (const float*)d_in[3];
  const float* b_hh   = (const float*)d_in[4];
  const float* fc1_w  = (const float*)d_in[5];
  const float* fc1_b  = (const float*)d_in[6];
  const float* fc2_w  = (const float*)d_in[7];
  const float* fc2_b  = (const float*)d_in[8];

  char* ws = (char*)d_ws;
  _Float16* Wcat  = (_Float16*)(ws + 0);         // 4,194,304 B
  float*    bias  = (float*)(ws + 4194304);      //     8,192 B
  _Float16* hbuf  = (_Float16*)(ws + 4202496);   //   524,288 B (2 phases)
  int*      done  = (int*)(ws + 4726784);        //     1,024 B
  int*      rdone = (int*)(ws + 4727808);        //     1,024 B

  prep_kernel<<<2178, 256, 0, stream>>>(W_ih, W_hh, b_ih, b_hh, Wcat, bias,
                                        hbuf, done, rdone);
  lstm_kernel<<<256, 128, LDS_BYTES, stream>>>(embeds, Wcat, bias, hbuf, done, rdone);
  head_kernel<<<256, 256, 0, stream>>>(hbuf /* final h is phase 0 */,
                                       fc1_w, fc1_b, fc2_w, fc2_b, (float*)d_out);
}
